// InferenceNetwork_45243185496666
// MI455X (gfx1250) — matrix operand order repaired and checked by reading, not compiled, not run on hardware
//
#include <hip/hip_runtime.h>

// ---- problem constants (from reference) ----
#define BB   1024   // batch
#define TT   10     // timesteps
#define HH   512    // hidden
#define DD   256    // embedding
#define PP   64     // pitch classes
#define INF_ 322    // P+2+D
#define KP   864    // fused K = 322+512=834 padded to mult of 32
#define NG   2048   // 4*H

typedef __attribute__((ext_vector_type(16))) _Float16 v16h;
typedef __attribute__((ext_vector_type(8)))  float    v8f;

#if defined(__HIP_DEVICE_COMPILE__) && __has_builtin(__builtin_amdgcn_tensor_load_to_lds)
#define HAVE_TDM 1
typedef __attribute__((ext_vector_type(4))) unsigned int u32x4_t;
typedef __attribute__((ext_vector_type(8))) int          i32x8_t;
typedef __attribute__((ext_vector_type(4))) int          i32x4_t;
#endif

struct __align__(16) H8 { _Float16 h[8]; };

__device__ __forceinline__ float sigm_(float x) { return 1.f / (1.f + expf(-x)); }

// Load a contiguous 16-f16 B-fragment (per-lane register order).
__device__ __forceinline__ v16h load_fragc(const _Float16* __restrict__ p)
{
    H8 x0 = *(const H8*)p;
    H8 x1 = *(const H8*)(p + 8);
    v16h v;
#pragma unroll
    for (int i = 0; i < 8; ++i) { v[i] = x0.h[i]; v[i + 8] = x1.h[i]; }
    return v;
}

// Build A-fragment from two contiguous 8-f16 chunks (K gap of 8 between them).
__device__ __forceinline__ v16h load_fraga(const _Float16* __restrict__ p)
{
    H8 x0 = *(const H8*)p;
    H8 x1 = *(const H8*)(p + 16);
    v16h v;
#pragma unroll
    for (int i = 0; i < 8; ++i) { v[i] = x0.h[i]; v[i + 8] = x1.h[i]; }
    return v;
}

#ifdef HAVE_TDM
// TDM: load a [rows x 4KB] B-slab (row stride rowStrideUnits*8 bytes) to LDS.
// 2-D descriptor, 8-byte units, per CDNA5 ISA.
__device__ __forceinline__ void tdm_load_slab(const _Float16* gsrc, unsigned lds_off,
                                              int rows, int rowStrideUnits)
{
    unsigned long long ga = (unsigned long long)(size_t)gsrc;
    u32x4_t g0;
    g0[0] = 1u;                                   // count=1, user descriptor
    g0[1] = lds_off;                              // lds_addr (bytes)
    g0[2] = (unsigned)(ga & 0xFFFFFFFFu);         // global_addr[31:0]
    g0[3] = (unsigned)((ga >> 32) & 0x01FFFFFFu)  // global_addr[56:32]
          | 0x80000000u;                          // type=2 (image)
    i32x8_t g1;
    g1[0] = (3 << 16);                            // data_size = 8B units
    g1[1] = (rowStrideUnits & 0xFFFF) << 16;      // tensor_dim0[15:0]
    g1[2] = ((rowStrideUnits >> 16) & 0xFFFF)     // tensor_dim0[31:16]
          | ((rows & 0xFFFF) << 16);              // tensor_dim1[15:0]
    g1[3] = (512 << 16);                          // tile_dim0 = 512 units (4KB)
    g1[4] = rows;                                 // tile_dim1
    g1[5] = rowStrideUnits;                       // tensor_dim0_stride[31:0]
    g1[6] = 0;
    g1[7] = 0;
    i32x4_t gz = {0, 0, 0, 0};
#if __clang_major__ >= 23
    i32x8_t gz8 = {0, 0, 0, 0, 0, 0, 0, 0};
    __builtin_amdgcn_tensor_load_to_lds(g0, g1, gz, gz, gz8, 0);
#else
    __builtin_amdgcn_tensor_load_to_lds(g0, g1, gz, gz, 0);
#endif
    __builtin_amdgcn_s_wait_tensorcnt(0);
}
#endif

// =====================================================================
// Weight packing: B-matrix fragments in exact per-lane register order.
// B frag (32x16 f16): lane n = lane&15, K = (lane>>4)*16 + e, e in [0,16).
// Packed layout: [kk][tn][lane][e] contiguous (512 f16 per 32x16 block).
// =====================================================================
__global__ void pack_lstm_w(const float* __restrict__ w_ih,  // (2048, 322)
                            const float* __restrict__ w_hh,  // (2048, 512)
                            _Float16* __restrict__ Wp)
{
    const int ntn = NG / 16;
    size_t idx = (size_t)blockIdx.x * blockDim.x + threadIdx.x;
    size_t total = (size_t)(KP / 32) * ntn * 512;
    if (idx >= total) return;
    int e    = idx & 15;
    int lane = (idx >> 4) & 31;
    size_t tb = idx >> 9;
    int tn = (int)(tb % ntn);
    int kk = (int)(tb / ntn);
    int n  = tn * 16 + (lane & 15);
    int k  = kk * 32 + (lane >> 4) * 16 + e;
    float v = 0.f;
    if (k < INF_)            v = w_ih[(size_t)n * INF_ + k];
    else if (k < INF_ + HH)  v = w_hh[(size_t)n * HH + (k - INF_)];
    Wp[idx] = (_Float16)v;
}

// Conv weights with tap-major K ordering: k = r*Cin + ci  (r = ky*3+kx).
__global__ void pack_conv_w(const float* __restrict__ w,  // (Cout, Cin, 3, 3)
                            _Float16* __restrict__ Wp, int Cout, int Cin)
{
    const int K = Cin * 9;
    const int ntn = Cout / 16;
    size_t idx = (size_t)blockIdx.x * blockDim.x + threadIdx.x;
    size_t total = (size_t)(K / 32) * ntn * 512;
    if (idx >= total) return;
    int e    = idx & 15;
    int lane = (idx >> 4) & 31;
    size_t tb = idx >> 9;
    int tn = (int)(tb % ntn);
    int kk = (int)(tb / ntn);
    int n  = tn * 16 + (lane & 15);
    int k  = kk * 32 + (lane >> 4) * 16 + e;
    int r  = k / Cin;          // filter tap
    int ci = k - r * Cin;      // input channel
    Wp[idx] = (_Float16)w[((size_t)n * Cin + ci) * 9 + r];
}

// =====================================================================
// conv1 (1->64ch, 3x3, pad 2, 28->30) + relu + maxpool(3,2) -> 14x14,
// written into a ZERO-PADDED NHWC buffer (B,16,16,64) for conv2 (pad=1).
// =====================================================================
__global__ void conv1_pool(const float* __restrict__ obs,
                           const float* __restrict__ w,   // (64,1,3,3)
                           const float* __restrict__ bias,
                           _Float16* __restrict__ out)    // NHWC (B,16,16,64)
{
    int idx = blockIdx.x * blockDim.x + threadIdx.x;
    int ch = idx & 63;  int t1 = idx >> 6;
    int xx = t1 & 15;   int t2 = t1 >> 4;
    int yy = t2 & 15;   int b  = t2 >> 4;
    if (b >= BB) return;
    int py = yy - 1, px = xx - 1;
    if (py < 0 || py >= 14 || px < 0 || px >= 14) { out[idx] = (_Float16)0.f; return; }

    float wr[9];
#pragma unroll
    for (int i = 0; i < 9; ++i) wr[i] = w[ch * 9 + i];
    float bv = bias[ch];
    const float* ob = obs + (size_t)b * 784;
    float mx = -1e30f;
#pragma unroll
    for (int wy = 0; wy < 3; ++wy)
#pragma unroll
        for (int wx = 0; wx < 3; ++wx) {
            int oy = py * 2 + wy, ox = px * 2 + wx;   // conv grid 30x30
            float s = bv;
#pragma unroll
            for (int ky = 0; ky < 3; ++ky)
#pragma unroll
                for (int kx = 0; kx < 3; ++kx) {
                    int iy = oy + ky - 2, ix = ox + kx - 2;
                    if (iy >= 0 && iy < 28 && ix >= 0 && ix < 28)
                        s += wr[ky * 3 + kx] * ob[iy * 28 + ix];
                }
            mx = fmaxf(mx, s);
        }
    out[idx] = (_Float16)fmaxf(mx, 0.f);
}

// =====================================================================
// Implicit-GEMM convolution via WMMA f16 -> f32, NHWC activations,
// tap-major K order. No bounds checks (input pre-padded / VALID).
// Block shares one 4-N-tile group; packed-B slab staged to LDS by TDM.
// Each wave: PM M-tiles x 4 N-tiles.
// =====================================================================
template <int CIN, int PM>
__global__ void __launch_bounds__(256, 2)
conv_gemm(const _Float16* __restrict__ in,   // NHWC (B,HinP,WinP,CIN)
          const _Float16* __restrict__ Wp,   // packed frags
          const float* __restrict__ bias,
          _Float16* __restrict__ out,        // NHWC (B,Hout,Wout,Cout)
          int HinP, int WinP,
          int Cout, int Hout, int Wout)
{
    int lane = threadIdx.x & 31;
    int wave = threadIdx.x >> 5;
    int ntn  = Cout >> 4;           // 16-wide N tiles
    int ng   = ntn >> 2;            // groups of 4 N tiles
    int tg   = blockIdx.x % ng;     // block-shared n-group
    int mb   = blockIdx.x / ng;
    int tm0  = (mb * 8 + wave) * PM;
    int hw   = Hout * Wout;
    int tn0  = tg * 4;
    const int nK = 9 * (CIN >> 5);

#ifdef HAVE_TDM
    extern __shared__ _Float16 ldsW[];      // nK * 2048 f16
    if (wave == 0)
        tdm_load_slab(Wp + (size_t)tn0 * 512, (unsigned)(size_t)(void*)ldsW,
                      nK, ntn * 128);
    __syncthreads();
#else
    const _Float16* wbase = Wp + ((size_t)tn0 * 512) + lane * 16;
#endif

    int kb = (lane >> 4) * 16;
    const _Float16* inb[PM];
#pragma unroll
    for (int p = 0; p < PM; ++p) {
        int m  = (tm0 + p) * 16 + (lane & 15);
        int bi = m / hw; int rem = m % hw;
        int oy = rem / Wout, ox = rem % Wout;
        inb[p] = in + (((size_t)bi * HinP + oy) * WinP + ox) * CIN + kb;
    }

    v8f acc[PM][4] = {};
#pragma unroll
    for (int r = 0; r < 9; ++r) {
        const int ky = r / 3, kx = r % 3;
        const size_t poff = ((size_t)ky * WinP + kx) * CIN;
#pragma unroll
        for (int cc = 0; cc < CIN; cc += 32) {
            v16h a[PM];
#pragma unroll
            for (int p = 0; p < PM; ++p) a[p] = load_fragc(inb[p] + poff + cc);
            int kk = r * (CIN >> 5) + (cc >> 5);
#ifdef HAVE_TDM
            const _Float16* wp = ldsW + kk * 2048 + lane * 16;
#else
            const _Float16* wp = wbase + (size_t)kk * ntn * 512;
#endif
            v16h b0 = load_fragc(wp);
#pragma unroll
            for (int p = 0; p < PM; ++p)
                acc[p][0] = __builtin_amdgcn_wmma_f32_16x16x32_f16(false, a[p], false, b0, (short)0, acc[p][0], false, false);
            v16h b1 = load_fragc(wp + 512);
#pragma unroll
            for (int p = 0; p < PM; ++p)
                acc[p][1] = __builtin_amdgcn_wmma_f32_16x16x32_f16(false, a[p], false, b1, (short)0, acc[p][1], false, false);
            v16h b2 = load_fragc(wp + 1024);
#pragma unroll
            for (int p = 0; p < PM; ++p)
                acc[p][2] = __builtin_amdgcn_wmma_f32_16x16x32_f16(false, a[p], false, b2, (short)0, acc[p][2], false, false);
            v16h b3 = load_fragc(wp + 1536);
#pragma unroll
            for (int p = 0; p < PM; ++p)
                acc[p][3] = __builtin_amdgcn_wmma_f32_16x16x32_f16(false, a[p], false, b3, (short)0, acc[p][3], false, false);
        }
    }

#pragma unroll
    for (int p = 0; p < PM; ++p) {
        int rb = (tm0 + p) * 16 + (lane >> 4) * 8;
#pragma unroll
        for (int j = 0; j < 4; ++j) {
            int co = (tn0 + j) * 16 + (lane & 15);
            float bv = bias[co];
#pragma unroll
            for (int v = 0; v < 8; ++v) {
                int mr = rb + v;
                int bb = mr / hw; int rr = mr % hw;
                int yy = rr / Wout, xx = rr % Wout;
                float val = fmaxf(acc[p][j][v] + bv, 0.f);
                out[(((size_t)bb * Hout + yy) * Wout + xx) * Cout + co] = (_Float16)val;
            }
        }
    }
}

// generic VALID maxpool on NHWC f16
__global__ void maxpool_nhwc(const _Float16* __restrict__ in, _Float16* __restrict__ out,
                             int C, int Hin, int Win, int Hout, int Wout,
                             int k, int s, int total)
{
    int idx = blockIdx.x * blockDim.x + threadIdx.x;
    if (idx >= total) return;
    int c  = idx % C;    int t1 = idx / C;
    int xx = t1 % Wout;  int t2 = t1 / Wout;
    int yy = t2 % Hout;  int b  = t2 / Hout;
    const _Float16* ib = in + ((size_t)b * Hin * Win) * C + c;
    float mx = -1e30f;
    for (int i = 0; i < k; ++i)
        for (int j = 0; j < k; ++j)
            mx = fmaxf(mx, (float)ib[((yy * s + i) * Win + (xx * s + j)) * C]);
    out[idx] = (_Float16)mx;
}

// conv4 out NHWC (B,2,2,256) -> maxpool 2x2 -> emb (B,256) f32
__global__ void pool_emb(const _Float16* __restrict__ c4, float* __restrict__ emb)
{
    int idx = blockIdx.x * blockDim.x + threadIdx.x;
    if (idx >= BB * DD) return;
    int c = idx & 255, b = idx >> 8;
    const _Float16* p = c4 + (size_t)b * 4 * DD + c;
    float mx = fmaxf(fmaxf((float)p[0], (float)p[DD]),
                     fmaxf((float)p[2 * DD], (float)p[3 * DD]));
    emb[idx] = mx;
}

__global__ void init_hc(float* __restrict__ h, float* __restrict__ c)
{
    int idx = blockIdx.x * blockDim.x + threadIdx.x;
    if (idx >= BB * HH) return;
    h[idx] = 0.f; c[idx] = 0.f;
}

// Build X_t = [onehot(ids[:,t-1]) | onehot(oo[:,t-1]) | emb | h | 0-pad] as f16
__global__ void pack_x(const int* __restrict__ ids, const int* __restrict__ oo,
                       const float* __restrict__ emb, const float* __restrict__ h,
                       _Float16* __restrict__ X, int t)
{
    int idx = blockIdx.x * blockDim.x + threadIdx.x;
    if (idx >= BB * KP) return;
    int k = idx % KP, b = idx / KP;
    float v = 0.f;
    if (k < PP + 2) {
        if (t > 0) {
            if (k < PP) v = (ids[b * TT + (t - 1)] == k) ? 1.f : 0.f;
            else        v = (oo [b * TT + (t - 1)] == (k - PP)) ? 1.f : 0.f;
        }
    } else if (k < INF_) {
        v = emb[(size_t)b * DD + (k - (PP + 2))];
    } else if (k < INF_ + HH) {
        v = h[(size_t)b * HH + (k - INF_)];
    }
    X[idx] = (_Float16)v;
}

// =====================================================================
// gates(B,2048) = X(B,864) @ Wfused(864,2048) + b_ih + b_hh.
// Block owns one 4-N-tile group; whole packed-B slab (27x4KB = 108KB)
// TDM-loaded to LDS once; 8 waves (2 M-tiles x 4 N-tiles each) consume.
// Grid: 128 blocks = 4 m-blocks x 32 n-groups.
// =====================================================================
__global__ void __launch_bounds__(256, 2)
gemm_gates(const _Float16* __restrict__ X,
           const _Float16* __restrict__ Wp,
           const float* __restrict__ b_ih,
           const float* __restrict__ b_hh,
           float* __restrict__ gates)
{
    const int ntn = NG / 16;   // 128 n-tiles
    int lane = threadIdx.x & 31;
    int wave = threadIdx.x >> 5;
    int tg  = blockIdx.x >> 2;          // n-group of 4 tiles
    int mb  = blockIdx.x & 3;           // m-block
    int tn0 = tg * 4;
    int tm0 = (mb * 8 + wave) * 2;      // 2 m-tiles per wave

#ifdef HAVE_TDM
    extern __shared__ _Float16 ldsW[];  // 27 * 2048 f16 = 110592 B
    if (wave == 0)
        tdm_load_slab(Wp + (size_t)tn0 * 512, (unsigned)(size_t)(void*)ldsW,
                      27, 16384);
    __syncthreads();
#else
    const _Float16* wbase = Wp + ((size_t)tn0 * 512) + lane * 16;
#endif

    int kb = (lane >> 4) * 16;
    const _Float16* xrow0 = X + (size_t)(tm0 * 16 + (lane & 15)) * KP + kb;
    const _Float16* xrow1 = xrow0 + (size_t)16 * KP;

    v8f a00 = {}, a01 = {}, a02 = {}, a03 = {};
    v8f a10 = {}, a11 = {}, a12 = {}, a13 = {};
    for (int kk = 0; kk < KP / 32; ++kk) {
        v16h af0 = load_fragc(xrow0 + kk * 32);
        v16h af1 = load_fragc(xrow1 + kk * 32);
#ifdef HAVE_TDM
        const _Float16* wp = ldsW + kk * 2048 + lane * 16;
#else
        const _Float16* wp = wbase + (size_t)kk * ntn * 512;
#endif
        v16h b0 = load_fragc(wp);
        a00 = __builtin_amdgcn_wmma_f32_16x16x32_f16(false, af0, false, b0, (short)0, a00, false, false);
        a10 = __builtin_amdgcn_wmma_f32_16x16x32_f16(false, af1, false, b0, (short)0, a10, false, false);
        v16h b1 = load_fragc(wp + 512);
        a01 = __builtin_amdgcn_wmma_f32_16x16x32_f16(false, af0, false, b1, (short)0, a01, false, false);
        a11 = __builtin_amdgcn_wmma_f32_16x16x32_f16(false, af1, false, b1, (short)0, a11, false, false);
        v16h b2 = load_fragc(wp + 1024);
        a02 = __builtin_amdgcn_wmma_f32_16x16x32_f16(false, af0, false, b2, (short)0, a02, false, false);
        a12 = __builtin_amdgcn_wmma_f32_16x16x32_f16(false, af1, false, b2, (short)0, a12, false, false);
        v16h b3 = load_fragc(wp + 1536);
        a03 = __builtin_amdgcn_wmma_f32_16x16x32_f16(false, af0, false, b3, (short)0, a03, false, false);
        a13 = __builtin_amdgcn_wmma_f32_16x16x32_f16(false, af1, false, b3, (short)0, a13, false, false);
    }

    v8f accs[2][4] = {{a00, a01, a02, a03}, {a10, a11, a12, a13}};
#pragma unroll
    for (int p = 0; p < 2; ++p) {
        int rb = (tm0 + p) * 16 + (lane >> 4) * 8;
#pragma unroll
        for (int j = 0; j < 4; ++j) {
            int col = (tn0 + j) * 16 + (lane & 15);
            float bias = b_ih[col] + b_hh[col];
#pragma unroll
            for (int v = 0; v < 8; ++v)
                gates[(size_t)(rb + v) * NG + col] = accs[p][j][v] + bias;
        }
    }
}

// LSTM cell + head + two log-softmaxes + log-prob accumulation. 1 block/sample.
__global__ void lstm_cell(const float* __restrict__ gates,
                          float* __restrict__ h, float* __restrict__ c,
                          const float* __restrict__ lin_w,  // (66,512)
                          const float* __restrict__ lin_b,  // (66,)
                          const int* __restrict__ ids, const int* __restrict__ oo,
                          float* __restrict__ out, int t)
{
    __shared__ float hs[HH];
    __shared__ float logits[PP + 2];
    int b = blockIdx.x;
    int j = threadIdx.x;

    const float* g = gates + (size_t)b * NG;
    float ig = g[j], fg = g[HH + j], gg = g[2 * HH + j], og = g[3 * HH + j];
    float cc = sigm_(fg) * c[(size_t)b * HH + j] + sigm_(ig) * tanhf(gg);
    float hh = sigm_(og) * tanhf(cc);
    c[(size_t)b * HH + j] = cc;
    h[(size_t)b * HH + j] = hh;
    hs[j] = hh;
    __syncthreads();

    if (j < PP + 2) {
        float acc = lin_b[j];
        const float* wr = lin_w + (size_t)j * HH;
        for (int i = 0; i < HH; ++i) acc += hs[i] * wr[i];
        logits[j] = acc;
    }
    __syncthreads();

    if (j == 0) {
        float mx = -1e30f;
        for (int i = 0; i < PP; ++i) mx = fmaxf(mx, logits[i]);
        float se = 0.f;
        for (int i = 0; i < PP; ++i) se += expf(logits[i] - mx);
        float lse = mx + logf(se);
        int idt = ids[b * TT + t];
        float id_lp = logits[idt] - lse;

        float m2 = fmaxf(logits[PP], logits[PP + 1]);
        float s2 = expf(logits[PP] - m2) + expf(logits[PP + 1] - m2);
        float lse2 = m2 + logf(s2);
        int oot = oo[b * TT + t];
        float oo_lp = logits[PP + oot] - lse2;

        float lp = id_lp + ((t > 0) ? 1.f : 0.f) * oo_lp;
        if (t == 0) out[b] = lp;
        else        out[b] += lp;
    }
}

// =====================================================================
extern "C" void kernel_launch(void* const* d_in, const int* in_sizes, int n_in,
                              void* d_out, int out_size, void* d_ws, size_t ws_size,
                              hipStream_t stream)
{
    const float* obs   = (const float*)d_in[0];
    const int*   ids   = (const int*)  d_in[1];
    const int*   oo    = (const int*)  d_in[2];
    const float* cw1   = (const float*)d_in[3];
    const float* cb1   = (const float*)d_in[4];
    const float* cw2   = (const float*)d_in[5];
    const float* cb2   = (const float*)d_in[6];
    const float* cw3   = (const float*)d_in[7];
    const float* cb3   = (const float*)d_in[8];
    const float* cw4   = (const float*)d_in[9];
    const float* cb4   = (const float*)d_in[10];
    const float* w_ih  = (const float*)d_in[11];
    const float* w_hh  = (const float*)d_in[12];
    const float* b_ih  = (const float*)d_in[13];
    const float* b_hh  = (const float*)d_in[14];
    const float* lin_w = (const float*)d_in[15];
    const float* lin_b = (const float*)d_in[16];
    float* out = (float*)d_out;

    char* ws = (char*)d_ws;
    size_t off = 0;
    auto alloc = [&](size_t bytes) {
        size_t o = off; off += (bytes + 255) & ~(size_t)255; return o;
    };

    _Float16* Wg  = (_Float16*)(ws + alloc((size_t)(KP / 32) * (NG / 16) * 512 * 2));
    _Float16* W2p = (_Float16*)(ws + alloc((size_t)18 * 8  * 512 * 2));
    _Float16* W3p = (_Float16*)(ws + alloc((size_t)36 * 8  * 512 * 2));
    _Float16* W4p = (_Float16*)(ws + alloc((size_t)36 * 16 * 512 * 2));
    _Float16* p1  = (_Float16*)(ws + alloc((size_t)BB * 16 * 16 * 64 * 2));  // padded NHWC
    _Float16* c2o = (_Float16*)(ws + alloc((size_t)BB * 14 * 14 * 128 * 2)); // NHWC
    _Float16* p2  = (_Float16*)(ws + alloc((size_t)BB * 6 * 6 * 128 * 2));   // NHWC
    _Float16* c3o = (_Float16*)(ws + alloc((size_t)BB * 4 * 4 * 128 * 2));   // NHWC
    _Float16* c4o = (_Float16*)(ws + alloc((size_t)BB * 2 * 2 * 256 * 2));   // NHWC
    float*    emb = (float*)   (ws + alloc((size_t)BB * DD * 4));
    _Float16* X   = (_Float16*)(ws + alloc((size_t)BB * KP * 2));
    float*    gts = (float*)   (ws + alloc((size_t)BB * NG * 4));
    float*    hb  = (float*)   (ws + alloc((size_t)BB * HH * 4));
    float*    cb  = (float*)   (ws + alloc((size_t)BB * HH * 4));

    // Allow large dynamic LDS for TDM-staged GEMMs (gfx1250: 320KB/WGP).
    hipFuncSetAttribute((const void*)gemm_gates,
                        hipFuncAttributeMaxDynamicSharedMemorySize, 27 * 4096);
    hipFuncSetAttribute(reinterpret_cast<const void*>(&conv_gemm<64, 2>),
                        hipFuncAttributeMaxDynamicSharedMemorySize, 18 * 4096);
    hipFuncSetAttribute(reinterpret_cast<const void*>(&conv_gemm<128, 2>),
                        hipFuncAttributeMaxDynamicSharedMemorySize, 36 * 4096);
    hipFuncSetAttribute(reinterpret_cast<const void*>(&conv_gemm<128, 1>),
                        hipFuncAttributeMaxDynamicSharedMemorySize, 36 * 4096);

    // ---- weight packing (once per launch) ----
    pack_lstm_w<<<(27 * 128 * 512 + 255) / 256, 256, 0, stream>>>(w_ih, w_hh, Wg);
    pack_conv_w<<<(18 * 8  * 512 + 255) / 256, 256, 0, stream>>>(cw2, W2p, 128, 64);
    pack_conv_w<<<(36 * 8  * 512 + 255) / 256, 256, 0, stream>>>(cw3, W3p, 128, 128);
    pack_conv_w<<<(36 * 16 * 512 + 255) / 256, 256, 0, stream>>>(cw4, W4p, 256, 128);

    // ---- CNN ----
    conv1_pool<<<(BB * 16 * 16 * 64 + 255) / 256, 256, 0, stream>>>(obs, cw1, cb1, p1);
    // conv2: M=200704, PM=2 -> 784 m-blocks * 2 n-groups = 1568 blocks, 72KB LDS
    conv_gemm<64, 2><<<1568, 256, 18 * 4096, stream>>>(p1, W2p, cb2, c2o, 16, 16, 128, 14, 14);
    maxpool_nhwc<<<(BB * 128 * 36 + 255) / 256, 256, 0, stream>>>(c2o, p2, 128, 14, 14, 6, 6, 3, 2, BB * 128 * 36);
    // conv3: M=16384, PM=2 -> 64 m-blocks * 2 = 128 blocks, 144KB LDS
    conv_gemm<128, 2><<<128, 256, 36 * 4096, stream>>>(p2, W3p, cb3, c3o, 6, 6, 128, 4, 4);
    // conv4: M=4096, PM=1 -> 32 m-blocks * 4 = 128 blocks, 144KB LDS
    conv_gemm<128, 1><<<128, 256, 36 * 4096, stream>>>(c3o, W4p, cb4, c4o, 4, 4, 256, 2, 2);
    pool_emb<<<(BB * DD + 255) / 256, 256, 0, stream>>>(c4o, emb);

    // ---- LSTM ----
    init_hc<<<(BB * HH + 255) / 256, 256, 0, stream>>>(hb, cb);
    for (int t = 0; t < TT; ++t) {
        pack_x<<<(BB * KP + 255) / 256, 256, 0, stream>>>(ids, oo, emb, hb, X, t);
        // gates: 4 m-blocks x 32 n-groups = 128 blocks, 108KB LDS each
        gemm_gates<<<128, 256, 27 * 4096, stream>>>(X, Wg, b_ih, b_hh, gts);
        lstm_cell<<<BB, HH, 0, stream>>>(gts, hb, cb, lin_w, lin_b, ids, oo, out, t);
    }
}
